// MultiHeadAttention_1142461301246
// MI455X (gfx1250) — compile-verified, matrix operand order checked by reading, not run on hardware
//
#include <hip/hip_runtime.h>
#include <hip/hip_bf16.h>

typedef __attribute__((ext_vector_type(16))) _Float16 v16h;
typedef __attribute__((ext_vector_type(8)))  _Float16 v8h;
typedef __attribute__((ext_vector_type(8)))  float    v8f;

static constexpr int SEQ = 2048;   // sequence length
static constexpr int DIM = 1024;   // model dim
static constexpr int NH  = 16;     // heads
static constexpr int HD  = 64;     // head dim

__device__ __forceinline__ v8f wmma_f16(v16h a, v16h b, v8f c) {
  // D = A(16x32 f16) * B(32x16 f16) + C(16x16 f32)
  return __builtin_amdgcn_wmma_f32_16x16x32_f16(false, a, false, b, (short)0, c,
                                                false, false);
}

// Loads the per-lane 16-half fragment: 8 contiguous halves at p, 8 at p+16.
// Caller has already applied the lane-half K offset (0 or 8).
__device__ __forceinline__ v16h load_frag(const _Float16* p) {
  v8h lo = *(const v8h*)(p);
  v8h hi = *(const v8h*)(p + 16);
  v16h r;
#pragma unroll
  for (int i = 0; i < 8; ++i) { r[i] = lo[i]; r[8 + i] = hi[i]; }
  return r;
}

// ---------------- prep kernels ----------------
// 8 contiguous f32 -> 8 f16 per thread (b128 in x2, b128 out).
__global__ void cvt_f16_kernel(const float* __restrict__ src,
                               _Float16* __restrict__ dst, int n) {
  int i = blockIdx.x * blockDim.x + threadIdx.x;
  int base = i * 8;
  if (base < n) {
    v8h d;
#pragma unroll
    for (int j = 0; j < 8; ++j) d[j] = (_Float16)src[base + j];
    *(v8h*)(dst + base) = d;
  }
}

// src [NH][DIM][HD] fp32 -> dst [NH][HD][DIM] f16 (transposed per head)
__global__ void transpose_w_kernel(const float* __restrict__ src,
                                   _Float16* __restrict__ dst) {
  int i = blockIdx.x * blockDim.x + threadIdx.x;
  if (i >= NH * DIM * HD) return;
  int h = i / (DIM * HD);
  int rem = i % (DIM * HD);
  int d = rem / HD;
  int k = rem % HD;
  dst[((size_t)h * HD + k) * DIM + d] = (_Float16)src[i];
}

// ---------------- projection GEMM: Q,K,V = X @ W ----------------
// One wave per (z, h, 16-row s-tile); computes all 4 n-tiles (HD=64).
// k-loop unrolled x2 with two fixed fragment register sets (no rotation movs).
__global__ __launch_bounds__(256) void proj_kernel(
    const _Float16* __restrict__ Xq, const _Float16* __restrict__ Xk,
    const _Float16* __restrict__ Xv, const _Float16* __restrict__ WqT,
    const _Float16* __restrict__ WkT, const _Float16* __restrict__ WvT,
    _Float16* __restrict__ Qh, _Float16* __restrict__ Kh,
    _Float16* __restrict__ VTh) {
  int wid  = blockIdx.x * (blockDim.x >> 5) + (threadIdx.x >> 5);
  int lane = threadIdx.x & 31;
  int z    = wid / (NH * (SEQ / 16));
  int rem  = wid % (NH * (SEQ / 16));
  int h    = rem / (SEQ / 16);
  int st   = rem % (SEQ / 16);
  const _Float16* X  = (z == 0) ? Xq : (z == 1) ? Xk : Xv;
  const _Float16* WT = (z == 0) ? WqT : (z == 1) ? WkT : WvT;

  int hi = lane >> 4, l15 = lane & 15, koff = hi * 8;
  int s0 = st * 16;
  v8f acc[4] = {};
  const _Float16* xrow = X + (size_t)(s0 + l15) * DIM + koff;
  const _Float16* wrow = WT + (size_t)h * HD * DIM + (size_t)l15 * DIM + koff;

  v16h a0 = load_frag(xrow);
  v16h b0[4];
#pragma unroll
  for (int nt = 0; nt < 4; ++nt) b0[nt] = load_frag(wrow + nt * 16 * DIM);

  for (int kb = 0; kb < DIM - 64; kb += 64) {
    v16h a1 = load_frag(xrow + kb + 32);
    v16h b1[4];
#pragma unroll
    for (int nt = 0; nt < 4; ++nt)
      b1[nt] = load_frag(wrow + nt * 16 * DIM + kb + 32);
#pragma unroll
    for (int nt = 0; nt < 4; ++nt) acc[nt] = wmma_f16(a0, b0[nt], acc[nt]);

    a0 = load_frag(xrow + kb + 64);
#pragma unroll
    for (int nt = 0; nt < 4; ++nt)
      b0[nt] = load_frag(wrow + nt * 16 * DIM + kb + 64);
#pragma unroll
    for (int nt = 0; nt < 4; ++nt) acc[nt] = wmma_f16(a1, b1[nt], acc[nt]);
  }
  {  // tail: kb = DIM-64 slice pair
    v16h a1 = load_frag(xrow + DIM - 32);
    v16h b1[4];
#pragma unroll
    for (int nt = 0; nt < 4; ++nt)
      b1[nt] = load_frag(wrow + nt * 16 * DIM + DIM - 32);
#pragma unroll
    for (int nt = 0; nt < 4; ++nt) acc[nt] = wmma_f16(a0, b0[nt], acc[nt]);
#pragma unroll
    for (int nt = 0; nt < 4; ++nt) acc[nt] = wmma_f16(a1, b1[nt], acc[nt]);
  }

  // C layout: element (m = r + 8*hi, n = l15) per n-tile
#pragma unroll
  for (int nt = 0; nt < 4; ++nt) {
#pragma unroll
    for (int r = 0; r < 8; ++r) {
      int s = s0 + r + 8 * hi;
      int n = nt * 16 + l15;
      _Float16 val = (_Float16)acc[nt][r];
      if (z == 0)      Qh[((size_t)h * SEQ + s) * HD + n] = val;
      else if (z == 1) Kh[((size_t)h * SEQ + s) * HD + n] = val;
      else             VTh[((size_t)h * HD + n) * SEQ + s] = val;  // V transposed
    }
  }
}

// ---------------- flash attention (causal) ----------------
// One wave per (h, 16-row q-tile); K/V stream staged global->LDS with the
// CDNA5 async engine (ASYNCcnt), double-buffered per wave.
static constexpr int AW = 4;  // waves per block in attention

__device__ __forceinline__ void async_prefetch_chunk(
    const _Float16* kbase, const _Float16* vtbase, int t0,
    _Float16* ldsbuf, int lane) {
  // K chunk: rows t0..t0+31 of [SEQ][HD] f16 -> one contiguous 4KB block.
  unsigned long long kg = (unsigned long long)(uintptr_t)(kbase + (size_t)t0 * HD);
  unsigned kl = (unsigned)(uintptr_t)ldsbuf;
#pragma unroll
  for (int j = 0; j < 8; ++j) {
    unsigned long long ga = kg + (unsigned)(j * 512 + lane * 16);
    unsigned la = kl + (unsigned)(j * 512 + lane * 16);
    asm volatile("global_load_async_to_lds_b128 %0, %1, off"
                 :: "v"(la), "v"(ga) : "memory");
  }
  // V chunk: 64 rows (dk) x 64B from VT (row stride SEQ halves).
  unsigned long long vg = (unsigned long long)(uintptr_t)(vtbase + t0);
  unsigned vl = (unsigned)(uintptr_t)(ldsbuf + 2048);
#pragma unroll
  for (int j = 0; j < 8; ++j) {
    int piece = j * 32 + lane;          // piece = row*4 + quarter
    int row = piece >> 2, q = piece & 3;
    unsigned long long ga = vg + (unsigned long long)row * (SEQ * 2) + (unsigned)(q * 16);
    unsigned la = vl + (unsigned)(piece * 16);
    asm volatile("global_load_async_to_lds_b128 %0, %1, off"
                 :: "v"(la), "v"(ga) : "memory");
  }
}

__global__ __launch_bounds__(AW * 32) void attn_kernel(
    const _Float16* __restrict__ Qh, const _Float16* __restrict__ Kh,
    const _Float16* __restrict__ VTh, _Float16* __restrict__ O16) {
  __shared__ _Float16 smem[AW][2][4096];  // per wave: 2 x (2048 K + 2048 V) halves
  int wv   = threadIdx.x >> 5;
  int wid  = blockIdx.x * AW + wv;
  int lane = threadIdx.x & 31;
  int h  = wid / (SEQ / 16);
  int qt = wid % (SEQ / 16);
  int hi = lane >> 4, l15 = lane & 15, koff = 8 * hi;
  int q0 = qt * 16;
  int qrow = q0 + l15;  // the q-row this lane owns for softmax stats

  // Q as the B operand of S^T = K @ Q^T  (lane = q column, contiguous load)
  const _Float16* qptr = Qh + ((size_t)h * SEQ + qrow) * HD;
  v16h bq0 = load_frag(qptr + 0 + koff);
  v16h bq1 = load_frag(qptr + 32 + koff);

  v8f acc[4] = {};
  float mrun = -1e30f, lrun = 0.f;
  const _Float16* kbase  = Kh + (size_t)h * SEQ * HD;
  const _Float16* vtbase = VTh + (size_t)h * HD * SEQ;
  int nch = (q0 + 16 + 31) / 32;  // 32-wide key chunks up to causal bound

  async_prefetch_chunk(kbase, vtbase, 0, &smem[wv][0][0], lane);

  for (int ch = 0; ch < nch; ++ch) {
    _Float16* buf = &smem[wv][ch & 1][0];
    if (ch + 1 < nch) {
      // The buffer we are about to overwrite was fully consumed last iter;
      // make sure its ds reads retired, then overlap next chunk's DMA.
      asm volatile("s_wait_dscnt 0x0" ::: "memory");
      async_prefetch_chunk(kbase, vtbase, (ch + 1) * 32,
                           &smem[wv][(ch + 1) & 1][0], lane);
      asm volatile("s_wait_asynccnt 0x10" ::: "memory");  // current chunk landed
    } else {
      asm volatile("s_wait_asynccnt 0x0" ::: "memory");
    }

    int t0 = ch * 32;
    // Two transposed 16x16 score tiles from LDS: rows t0.. / t0+16..
    const _Float16* k0 = buf + (size_t)l15 * HD + koff;         // tile0 rows
    const _Float16* k1 = buf + (size_t)(16 + l15) * HD + koff;  // tile1 rows
    v8f st0 = {}, st1 = {};
    st0 = wmma_f16(load_frag(k0),      bq0, st0);
    st0 = wmma_f16(load_frag(k0 + 32), bq1, st0);
    st1 = wmma_f16(load_frag(k1),      bq0, st1);
    st1 = wmma_f16(load_frag(k1 + 32), bq1, st1);

    // scale (1/sqrt(64)), causal mask, chunk row-max (half-wave reduce)
    float cm = -1e30f;
    float p0[8], p1[8];
#pragma unroll
    for (int r = 0; r < 8; ++r) {
      int ta = t0 + r + 8 * hi;
      int tb = ta + 16;
      float sa = st0[r] * 0.125f;
      float sb = st1[r] * 0.125f;
      sa = (ta <= qrow) ? sa : -1e30f;
      sb = (tb <= qrow) ? sb : -1e30f;
      p0[r] = sa; p1[r] = sb;
      cm = fmaxf(cm, fmaxf(sa, sb));
    }
    cm = fmaxf(cm, __shfl_xor(cm, 16, 32));
    float mnew   = fmaxf(mrun, cm);
    float scalef = __expf(mrun - mnew);

    // exp + row-sum; pack P directly into the A-fragment layout (lane-local!)
    float rs = 0.f;
    v16h pa;
#pragma unroll
    for (int r = 0; r < 8; ++r) {
      float ea = __expf(p0[r] - mnew);
      float eb = __expf(p1[r] - mnew);
      rs += ea + eb;
      pa[r]     = (_Float16)ea;
      pa[8 + r] = (_Float16)eb;
    }
    rs += __shfl_xor(rs, 16, 32);
    lrun = lrun * scalef + rs;
    mrun = mnew;

    // rescale accumulator: per-row factors gathered from owning lanes
    float f[8];
#pragma unroll
    for (int r = 0; r < 8; ++r) f[r] = __shfl(scalef, r + 8 * hi, 32);
#pragma unroll
    for (int nt = 0; nt < 4; ++nt)
#pragma unroll
      for (int r = 0; r < 8; ++r) acc[nt][r] *= f[r];

    // O += P @ V ; V fragments are contiguous rows of the LDS-resident VT chunk
    const _Float16* vbuf = buf + 2048;
#pragma unroll
    for (int nt = 0; nt < 4; ++nt) {
      v16h bv = load_frag(vbuf + (size_t)(nt * 16 + l15) * 32 + koff);
      acc[nt] = wmma_f16(pa, bv, acc[nt]);
    }
  }

  // normalize and store concat-head output (f16 for the final GEMM)
  float linv = 1.0f / lrun;
  float f[8];
#pragma unroll
  for (int r = 0; r < 8; ++r) f[r] = __shfl(linv, r + 8 * hi, 32);
#pragma unroll
  for (int nt = 0; nt < 4; ++nt) {
#pragma unroll
    for (int r = 0; r < 8; ++r) {
      int s  = q0 + r + 8 * hi;
      int dk = nt * 16 + l15;
      O16[(size_t)s * DIM + h * HD + dk] = (_Float16)(acc[nt][r] * f[r]);
    }
  }
}

// ---------------- output projection: out = O @ Wo^T (fp32 out) ----------------
__global__ __launch_bounds__(256) void outproj_kernel(
    const _Float16* __restrict__ O16, const _Float16* __restrict__ Wo16,
    float* __restrict__ out) {
  int wid  = blockIdx.x * (blockDim.x >> 5) + (threadIdx.x >> 5);
  int lane = threadIdx.x & 31;
  int mt = wid / (DIM / HD);     // 128 s-tiles
  int ng = wid % (DIM / HD);     // 16 groups of 64 output cols
  int hi = lane >> 4, l15 = lane & 15, koff = 8 * hi;
  int s0 = mt * 16, e0 = ng * 64;

  const _Float16* orow = O16 + (size_t)(s0 + l15) * DIM + koff;
  const _Float16* wrow = Wo16 + (size_t)(e0 + l15) * DIM + koff;  // Wo is [out,in]
  v8f acc[4] = {};
  v16h a0 = load_frag(orow);
  v16h b0[4];
#pragma unroll
  for (int nt = 0; nt < 4; ++nt) b0[nt] = load_frag(wrow + (size_t)nt * 16 * DIM);

  for (int kb = 0; kb < DIM - 64; kb += 64) {
    v16h a1 = load_frag(orow + kb + 32);
    v16h b1[4];
#pragma unroll
    for (int nt = 0; nt < 4; ++nt)
      b1[nt] = load_frag(wrow + (size_t)nt * 16 * DIM + kb + 32);
#pragma unroll
    for (int nt = 0; nt < 4; ++nt) acc[nt] = wmma_f16(a0, b0[nt], acc[nt]);

    a0 = load_frag(orow + kb + 64);
#pragma unroll
    for (int nt = 0; nt < 4; ++nt)
      b0[nt] = load_frag(wrow + (size_t)nt * 16 * DIM + kb + 64);
#pragma unroll
    for (int nt = 0; nt < 4; ++nt) acc[nt] = wmma_f16(a1, b1[nt], acc[nt]);
  }
  {  // tail
    v16h a1 = load_frag(orow + DIM - 32);
    v16h b1[4];
#pragma unroll
    for (int nt = 0; nt < 4; ++nt)
      b1[nt] = load_frag(wrow + (size_t)nt * 16 * DIM + DIM - 32);
#pragma unroll
    for (int nt = 0; nt < 4; ++nt) acc[nt] = wmma_f16(a0, b0[nt], acc[nt]);
#pragma unroll
    for (int nt = 0; nt < 4; ++nt) acc[nt] = wmma_f16(a1, b1[nt], acc[nt]);
  }

#pragma unroll
  for (int nt = 0; nt < 4; ++nt) {
#pragma unroll
    for (int r = 0; r < 8; ++r) {
      int s = s0 + r + 8 * hi;
      int e = e0 + nt * 16 + l15;
      out[(size_t)s * DIM + e] = acc[nt][r];
    }
  }
}

// ---------------- host launch ----------------
extern "C" void kernel_launch(void* const* d_in, const int* in_sizes, int n_in,
                              void* d_out, int out_size, void* d_ws, size_t ws_size,
                              hipStream_t stream) {
  const float* q_f32  = (const float*)d_in[0];
  const float* k_f32  = (const float*)d_in[1];
  const float* v_f32  = (const float*)d_in[2];
  // d_in[3] = mask (int), unused: causal mask computed analytically
  const float* wq_f32 = (const float*)d_in[4];
  const float* wk_f32 = (const float*)d_in[5];
  const float* wv_f32 = (const float*)d_in[6];
  const float* wo_f32 = (const float*)d_in[7];
  float* out = (float*)d_out;

  const size_t SD2 = (size_t)SEQ * DIM * sizeof(_Float16);        // 4 MB
  const size_t WH2 = (size_t)NH * HD * DIM * sizeof(_Float16);    // 2 MB
  char* w = (char*)d_ws;
  _Float16* Xq16 = (_Float16*)(w);
  _Float16* Xk16 = (_Float16*)(w + SD2);
  _Float16* Xv16 = (_Float16*)(w + 2 * SD2);
  _Float16* WqT  = (_Float16*)(w + 3 * SD2);
  _Float16* WkT  = (_Float16*)(w + 3 * SD2 + WH2);
  _Float16* WvT  = (_Float16*)(w + 3 * SD2 + 2 * WH2);
  _Float16* Wo16 = (_Float16*)(w + 3 * SD2 + 3 * WH2);
  _Float16* Q16  = (_Float16*)(w + 4 * SD2 + 3 * WH2);  // Wo16 is DIM*DIM*2 = 2 MB, in an SD2 slot
  _Float16* K16  = (_Float16*)(w + 5 * SD2 + 3 * WH2);
  _Float16* VT16 = (_Float16*)(w + 6 * SD2 + 3 * WH2);
  _Float16* O16  = (_Float16*)(w + 7 * SD2 + 3 * WH2);

  const int nX  = SEQ * DIM;           // 2,097,152
  const int nW  = NH * DIM * HD;       // 1,048,576
  const int nWo = DIM * DIM;           // 1,048,576
  cvt_f16_kernel<<<(nX / 8 + 255) / 256, 256, 0, stream>>>(q_f32, Xq16, nX);
  cvt_f16_kernel<<<(nX / 8 + 255) / 256, 256, 0, stream>>>(k_f32, Xk16, nX);
  cvt_f16_kernel<<<(nX / 8 + 255) / 256, 256, 0, stream>>>(v_f32, Xv16, nX);
  transpose_w_kernel<<<(nW + 255) / 256, 256, 0, stream>>>(wq_f32, WqT);
  transpose_w_kernel<<<(nW + 255) / 256, 256, 0, stream>>>(wk_f32, WkT);
  transpose_w_kernel<<<(nW + 255) / 256, 256, 0, stream>>>(wv_f32, WvT);
  cvt_f16_kernel<<<(nWo / 8 + 255) / 256, 256, 0, stream>>>(wo_f32, Wo16, nWo);

  // 3 * 16 heads * 128 s-tiles = 6144 waves, 8 waves/block
  proj_kernel<<<768, 256, 0, stream>>>(Xq16, Xk16, Xv16, WqT, WkT, WvT,
                                       Q16, K16, VT16);
  // 16 heads * 128 q-tiles = 2048 waves, AW waves/block
  attn_kernel<<<2048 / AW, AW * 32, 0, stream>>>(Q16, K16, VT16, O16);
  // 128 s-tiles * 16 col-groups = 2048 waves
  outproj_kernel<<<256, 256, 0, stream>>>(O16, Wo16, out);
}
